// HybridMoE_20839181320753
// MI455X (gfx1250) — matrix operand order repaired and checked by reading, not compiled, not run on hardware
//
#include <hip/hip_runtime.h>
#include <hip/hip_bf16.h>
#include <stdint.h>

#define TT 2048
#define HH 2048
#define II 1408
#define EE 8
#define KKEXP 2

typedef __attribute__((ext_vector_type(16))) __bf16 v16bf;
typedef __attribute__((ext_vector_type(8)))  float  v8f;
typedef __attribute__((ext_vector_type(4)))  unsigned int u32x4;

union BFrag { u32x4 q[2]; v16bf v; };

#if __has_builtin(__builtin_amdgcn_cvt_pk_bf16_f32)
__device__ __forceinline__ unsigned int pack_bf16(float lo, float hi) {
  union { decltype(__builtin_amdgcn_cvt_pk_bf16_f32(0.0f, 0.0f)) v; unsigned int u; } c;
  c.v = __builtin_amdgcn_cvt_pk_bf16_f32(lo, hi);
  return c.u;
}
#else
__device__ __forceinline__ unsigned int pack_bf16(float lo, float hi) {
  unsigned int a = __float_as_uint(lo);
  unsigned int b = __float_as_uint(hi);
  a += 0x7fffu + ((a >> 16) & 1u);       // RNE
  b += 0x7fffu + ((b >> 16) & 1u);
  return __builtin_amdgcn_perm(b, a, 0x07060302u);  // {b[31:16], a[31:16]}
}
#endif

__device__ __forceinline__ unsigned short bf16_1(float x) {
  unsigned int a = __float_as_uint(x);
  a += 0x7fffu + ((a >> 16) & 1u);
  return (unsigned short)(a >> 16);
}

// Flat shared-pointer low 32 bits == LDS address consumed by async-to-LDS ops
// (ISA 10.2: LDS aperture mapping truncates addr to [31:0]).
__device__ __forceinline__ unsigned int lds_addr32(const void* p) {
  return (unsigned int)(unsigned long long)(uintptr_t)p;
}

__device__ __forceinline__ void async_load_b128(unsigned int lds_off, const void* gp) {
  asm volatile("global_load_async_to_lds_b128 %0, %1, off"
               :: "v"(lds_off), "v"((unsigned long long)(uintptr_t)gp)
               : "memory");
}
__device__ __forceinline__ void wait_asynccnt0() {
  asm volatile("s_wait_asynccnt 0" ::: "memory");
}

// ---------------------------------------------------------------- routing
__global__ void moe_route(const float* __restrict__ logits,
                          float* __restrict__ wts, int* __restrict__ cnt,
                          int* __restrict__ list) {
  int t = blockIdx.x * blockDim.x + threadIdx.x;
  if (t >= TT) return;
  float v[EE];
#pragma unroll
  for (int e = 0; e < EE; ++e) v[e] = logits[t * EE + e];
  int i0 = 0; float v0 = v[0];
#pragma unroll
  for (int e = 1; e < EE; ++e) if (v[e] > v0) { v0 = v[e]; i0 = e; }
  int i1 = -1; float v1 = -3.4e38f;
#pragma unroll
  for (int e = 0; e < EE; ++e) if (e != i0 && v[e] > v1) { v1 = v[e]; i1 = e; }
  float ex = __expf(v1 - v0);                  // softmax over top-2
  float w0 = 1.0f / (1.0f + ex);
  wts[t * 2 + 0] = w0;
  wts[t * 2 + 1] = 1.0f - w0;
  int p0 = atomicAdd(&cnt[i0], 1);
  list[i0 * TT + p0] = t * 2 + 0;
  int p1 = atomicAdd(&cnt[i1], 1);
  list[i1 * TT + p1] = t * 2 + 1;
}

// ---------------------------------------------------------------- gate+up
// 64 expert rows x 64 I cols per block, K = H, double-buffered LDS
__global__ __launch_bounds__(256)
void moe_gateup(const float* __restrict__ X,
                const float* __restrict__ Wg, const float* __restrict__ Wu,
                const int* __restrict__ cnt, const int* __restrict__ list,
                unsigned short* __restrict__ act) {
  const int e  = blockIdx.z;
  const int n0 = blockIdx.x * 64;
  const int m0 = blockIdx.y * 64;
  const int ce = cnt[e];
  if (m0 >= ce) return;

  __shared__ unsigned int sX[2][64 * 16];     // 64 rows x 32 bf16
  __shared__ unsigned int sWg[2][4 * 32 * 8]; // B-fragment order
  __shared__ unsigned int sWu[2][4 * 32 * 8];
  __shared__ int sAid[64];

  const int tid = threadIdx.x;
  if (tid < 64) {
    int m = m0 + tid;
    sAid[tid] = (m < ce) ? list[e * TT + m] : -1;
  }
  __syncthreads();

  const float* wg = Wg + (size_t)e * HH * II;
  const float* wu = Wu + (size_t)e * HH * II;

  const int lane = tid & 31;
  const int wave = tid >> 5;
  const int rg = wave >> 1;
  const int cg = wave & 1;

  v8f accg0 = {}, accg1 = {}, accu0 = {}, accu1 = {};

  // ---- per-thread staging bases (running pointers, no per-iter muls) ----
  const int xr = tid >> 2;
  const int xs = tid & 3;
  const int tok0 = (sAid[xr] >= 0) ? (sAid[xr] >> 1) : 0;   // clamp: garbage rows masked at store
  const float* xp = X + (size_t)tok0 * HH + xs * 8;

  const int k0 = (tid >> 5) * 2;              // 0..14 even
  const int nn = (tid & 31) * 2;              // 0..62 even
  const int wbase = (((nn >> 4) * 32) + (nn & 15)) * 8 + (k0 >> 1);  // micro1 = +128
  const float* wgp = wg + (size_t)k0 * II + n0 + nn;
  const float* wup = wu + (size_t)k0 * II + n0 + nn;
  const float* pfg = wg + (size_t)((tid >> 3) + 64) * II + n0 + (tid & 7) * 8;
  const float* pfu = wu + (size_t)((tid >> 3) + 64) * II + n0 + (tid & 7) * 8;

  auto stage = [&](int b) {
    float4 f0 = *(const float4*)(xp);
    float4 f1 = *(const float4*)(xp + 4);
    u32x4 p;
    p.x = pack_bf16(f0.x, f0.y); p.y = pack_bf16(f0.z, f0.w);
    p.z = pack_bf16(f1.x, f1.y); p.w = pack_bf16(f1.z, f1.w);
    *(u32x4*)&sX[b][xr * 16 + xs * 4] = p;

    float2 g0 = *(const float2*)(wgp);
    float2 g1 = *(const float2*)(wgp + II);
    float2 g2 = *(const float2*)(wgp + 16 * II);
    float2 g3 = *(const float2*)(wgp + 17 * II);
    sWg[b][wbase]       = pack_bf16(g0.x, g1.x);
    sWg[b][wbase + 8]   = pack_bf16(g0.y, g1.y);
    sWg[b][wbase + 128] = pack_bf16(g2.x, g3.x);
    sWg[b][wbase + 136] = pack_bf16(g2.y, g3.y);

    float2 u0 = *(const float2*)(wup);
    float2 u1 = *(const float2*)(wup + II);
    float2 u2 = *(const float2*)(wup + 16 * II);
    float2 u3 = *(const float2*)(wup + 17 * II);
    sWu[b][wbase]       = pack_bf16(u0.x, u1.x);
    sWu[b][wbase + 8]   = pack_bf16(u0.y, u1.y);
    sWu[b][wbase + 128] = pack_bf16(u2.x, u3.x);
    sWu[b][wbase + 136] = pack_bf16(u2.y, u3.y);

    xp += 32;
    wgp += (size_t)32 * II;
    wup += (size_t)32 * II;
  };

  stage(0);
  int cur = 0;
  for (int kk = 0; kk < HH; kk += 32) {
    __syncthreads();                  // staged buffer `cur` now visible
    if (kk + 32 < HH) {
      stage(cur ^ 1);
      if (kk + 64 < HH) {             // prefetch two steps ahead
        __builtin_prefetch(pfg, 0, 1);
        __builtin_prefetch(pfu, 0, 1);
        pfg += (size_t)32 * II;
        pfu += (size_t)32 * II;
      }
    }
    const unsigned int* bX = sX[cur];
    const unsigned int* bG = sWg[cur];
    const unsigned int* bU = sWu[cur];
    // load A + all 4 B fragments into distinct regs, then issue WMMAs
    BFrag a, b0, b1, b2, b3;
    {
      int row = rg * 16 + (lane & 15);
      int half = (lane >> 4) & 1;
      a.q[0] = *(const u32x4*)&bX[row * 16 + half * 8];
      a.q[1] = *(const u32x4*)&bX[row * 16 + half * 8 + 4];
    }
    {
      int acg = cg * 2;
      b0.q[0] = *(const u32x4*)&bG[(acg * 32 + lane) * 8];
      b0.q[1] = *(const u32x4*)&bG[(acg * 32 + lane) * 8 + 4];
      b1.q[0] = *(const u32x4*)&bG[((acg + 1) * 32 + lane) * 8];
      b1.q[1] = *(const u32x4*)&bG[((acg + 1) * 32 + lane) * 8 + 4];
      b2.q[0] = *(const u32x4*)&bU[(acg * 32 + lane) * 8];
      b2.q[1] = *(const u32x4*)&bU[(acg * 32 + lane) * 8 + 4];
      b3.q[0] = *(const u32x4*)&bU[((acg + 1) * 32 + lane) * 8];
      b3.q[1] = *(const u32x4*)&bU[((acg + 1) * 32 + lane) * 8 + 4];
    }
    // scheduling fence: force all fragments live -> one dscnt drain, WMMAs back-to-back
    asm volatile("" : "+v"(a.q[0]), "+v"(a.q[1]),
                      "+v"(b0.q[0]), "+v"(b0.q[1]), "+v"(b1.q[0]), "+v"(b1.q[1]),
                      "+v"(b2.q[0]), "+v"(b2.q[1]), "+v"(b3.q[0]), "+v"(b3.q[1]));
    accg0 = __builtin_amdgcn_wmma_f32_16x16x32_bf16(false, a.v, false, b0.v, (short)0, accg0, false, false);
    accg1 = __builtin_amdgcn_wmma_f32_16x16x32_bf16(false, a.v, false, b1.v, (short)0, accg1, false, false);
    accu0 = __builtin_amdgcn_wmma_f32_16x16x32_bf16(false, a.v, false, b2.v, (short)0, accu0, false, false);
    accu1 = __builtin_amdgcn_wmma_f32_16x16x32_bf16(false, a.v, false, b3.v, (short)0, accu1, false, false);
    cur ^= 1;
  }

  const int colb  = n0 + cg * 32 + (lane & 15);
  const int rbase = rg * 16 + ((lane >> 4) ? 8 : 0);
#pragma unroll
  for (int i = 0; i < 8; ++i) {
    int aid = sAid[rbase + i];
    if (aid >= 0) {
      float g0 = accg0[i], g1 = accg1[i];
      float a0 = (g0 / (1.0f + __expf(-g0))) * accu0[i];
      float a1 = (g1 / (1.0f + __expf(-g1))) * accu1[i];
      act[(size_t)aid * II + colb]      = bf16_1(a0);
      act[(size_t)aid * II + colb + 16] = bf16_1(a1);
    }
  }
}

// ---------------------------------------------------------------- down proj
// A-tile staged with GLOBAL_LOAD_ASYNC_TO_LDS_B128 (pure bf16 copy)
__global__ __launch_bounds__(256)
void moe_down(const unsigned short* __restrict__ act,
              const float* __restrict__ Wd,
              const int* __restrict__ cnt, const int* __restrict__ list,
              const float* __restrict__ wts,
              float* __restrict__ ybuf) {
  const int e  = blockIdx.z;
  const int n0 = blockIdx.x * 64;
  const int m0 = blockIdx.y * 64;
  const int ce = cnt[e];
  if (m0 >= ce) return;

  __shared__ unsigned int sA[2][64 * 16];
  __shared__ unsigned int sW[2][4 * 32 * 8];
  __shared__ int sAid[64];
  __shared__ float sWt[64];

  const int tid = threadIdx.x;
  if (tid < 64) {
    int m = m0 + tid;
    int aid = (m < ce) ? list[e * TT + m] : -1;
    sAid[tid] = aid;
    sWt[tid] = (aid >= 0) ? wts[aid] : 0.0f;
  }
  __syncthreads();

  const float* wd = Wd + (size_t)e * II * HH;

  const int lane = tid & 31;
  const int wave = tid >> 5;
  const int rg = wave >> 1;
  const int cg = wave & 1;

  v8f acc0 = {}, acc1 = {};

  const int xr = tid >> 2;
  const int xs = tid & 3;
  const int arow0 = sAid[xr];
  const int arowc = (arow0 >= 0) ? arow0 : 0;   // clamp: in-bounds garbage, masked at store
  const unsigned short* agp = act + (size_t)arowc * II + xs * 8;
  const unsigned int ldsA0 = lds_addr32(&sA[0][xr * 16 + xs * 4]);
  const unsigned int ldsA1 = lds_addr32(&sA[1][xr * 16 + xs * 4]);

  const int k0 = (tid >> 5) * 2;
  const int nn = (tid & 31) * 2;
  const int wbase = (((nn >> 4) * 32) + (nn & 15)) * 8 + (k0 >> 1);
  const float* wdp = wd + (size_t)k0 * HH + n0 + nn;
  const float* pfd = wd + (size_t)((tid >> 3) + 64) * HH + n0 + (tid & 7) * 8;

  auto stageA = [&](int b) {
    async_load_b128(b ? ldsA1 : ldsA0, agp);
    agp += 32;
  };
  auto stageW = [&](int b) {
    float2 r0 = *(const float2*)(wdp);
    float2 r1 = *(const float2*)(wdp + HH);
    float2 r2 = *(const float2*)(wdp + 16 * HH);
    float2 r3 = *(const float2*)(wdp + 17 * HH);
    sW[b][wbase]       = pack_bf16(r0.x, r1.x);
    sW[b][wbase + 8]   = pack_bf16(r0.y, r1.y);
    sW[b][wbase + 128] = pack_bf16(r2.x, r3.x);
    sW[b][wbase + 136] = pack_bf16(r2.y, r3.y);
    wdp += (size_t)32 * HH;
  };

  stageA(0); stageW(0);
  int cur = 0;
  for (int kk = 0; kk < II; kk += 32) {
    wait_asynccnt0();                 // buffer `cur` async data landed in LDS
    __syncthreads();
    if (kk + 32 < II) {
      stageA(cur ^ 1);
      stageW(cur ^ 1);
      if (kk + 64 < II) {
        __builtin_prefetch(pfd, 0, 1);
        pfd += (size_t)32 * HH;
      }
    }
    const unsigned int* bA = sA[cur];
    const unsigned int* bW = sW[cur];
    BFrag a, b0, b1;
    {
      int row = rg * 16 + (lane & 15);
      int half = (lane >> 4) & 1;
      a.q[0] = *(const u32x4*)&bA[row * 16 + half * 8];
      a.q[1] = *(const u32x4*)&bA[row * 16 + half * 8 + 4];
    }
    {
      int acg = cg * 2;
      b0.q[0] = *(const u32x4*)&bW[(acg * 32 + lane) * 8];
      b0.q[1] = *(const u32x4*)&bW[(acg * 32 + lane) * 8 + 4];
      b1.q[0] = *(const u32x4*)&bW[((acg + 1) * 32 + lane) * 8];
      b1.q[1] = *(const u32x4*)&bW[((acg + 1) * 32 + lane) * 8 + 4];
    }
    asm volatile("" : "+v"(a.q[0]), "+v"(a.q[1]),
                      "+v"(b0.q[0]), "+v"(b0.q[1]), "+v"(b1.q[0]), "+v"(b1.q[1]));
    acc0 = __builtin_amdgcn_wmma_f32_16x16x32_bf16(false, a.v, false, b0.v, (short)0, acc0, false, false);
    acc1 = __builtin_amdgcn_wmma_f32_16x16x32_bf16(false, a.v, false, b1.v, (short)0, acc1, false, false);
    cur ^= 1;
  }

  const int colb  = n0 + cg * 32 + (lane & 15);
  const int rbase = rg * 16 + ((lane >> 4) ? 8 : 0);
#pragma unroll
  for (int i = 0; i < 8; ++i) {
    int rl = rbase + i;
    int aid = sAid[rl];
    if (aid >= 0) {
      float w = sWt[rl];
      ybuf[(size_t)aid * HH + colb]      = acc0[i] * w;
      ybuf[(size_t)aid * HH + colb + 16] = acc1[i] * w;
    }
  }
}

// ---------------------------------------------------------------- combine
__global__ void moe_combine(const float* __restrict__ ybuf, float* __restrict__ out) {
  int i = blockIdx.x * blockDim.x + threadIdx.x;
  int o = i * 4;
  int t = o >> 11;
  int h = o & (HH - 1);
  const float4 a = *(const float4*)&ybuf[(size_t)(t * 2) * HH + h];
  const float4 b = *(const float4*)&ybuf[(size_t)(t * 2 + 1) * HH + h];
  float4 r; r.x = a.x + b.x; r.y = a.y + b.y; r.z = a.z + b.z; r.w = a.w + b.w;
  *(float4*)&out[o] = r;
}

extern "C" void kernel_launch(void* const* d_in, const int* in_sizes, int n_in,
                              void* d_out, int out_size, void* d_ws, size_t ws_size,
                              hipStream_t stream) {
  (void)in_sizes; (void)n_in; (void)out_size; (void)ws_size;
  const float* X  = (const float*)d_in[0];
  const float* rl = (const float*)d_in[1];
  const float* Wg = (const float*)d_in[2];
  const float* Wu = (const float*)d_in[3];
  const float* Wd = (const float*)d_in[4];

  char* ws = (char*)d_ws;
  float*          wts  = (float*)(ws);
  int*            cnt  = (int*)(ws + 16384);
  int*            list = (int*)(ws + 32768);
  unsigned short* act  = (unsigned short*)(ws + 98304);
  float*          ybuf = (float*)(ws + 98304 + (size_t)TT * KKEXP * II * 2);

  hipMemsetAsync(cnt, 0, EE * sizeof(int), stream);
  moe_route<<<TT / 256, 256, 0, stream>>>(rl, wts, cnt, list);
  moe_gateup<<<dim3(II / 64, TT / 64, EE), 256, 0, stream>>>(X, Wg, Wu, cnt, list, act);
  moe_down<<<dim3(HH / 64, TT / 64, EE), 256, 0, stream>>>(act, Wd, cnt, list, wts, ybuf);
  moe_combine<<<(TT * HH / 4) / 256, 256, 0, stream>>>(ybuf, (float*)d_out);
}
